// GraphTransformer_28467043238278
// MI455X (gfx1250) — matrix operand order repaired and checked by reading, not compiled, not run on hardware
//
#include <hip/hip_runtime.h>
#include <hip/hip_bf16.h>
#include <math.h>

// ---------------------------------------------------------------------------
// GraphTransformer (PyG TransformerConv, beta=True) for MI455X / gfx1250.
// Dense GEMMs: v_wmma_f32_16x16x32_bf16, bf16 operands pre-converted once
// (weights) or produced fused in the previous kernel's epilogue (activations),
// so the WMMA inner loop is pure global_load_b128 + wmma.
// ---------------------------------------------------------------------------

#define HID 128
#define NHEAD 8
#define DHEAD 16

typedef __bf16 v16bf __attribute__((ext_vector_type(16)));
typedef float  v8f   __attribute__((ext_vector_type(8)));

union V16 { v16bf v; uint4 u[2]; };   // 32 bytes: 16 bf16 <-> two 16B chunks

__device__ __forceinline__ float gelu_exact(float x) {
    return 0.5f * x * (1.0f + erff(x * 0.70710678118654752f));
}

__device__ __forceinline__ void atomicMaxF(float* addr, float val) {
    int* ia = (int*)addr;
    int cur = __float_as_int(*addr);
    while (__int_as_float(cur) < val) {
        int prev = atomicCAS(ia, cur, __float_as_int(val));
        if (prev == cur) break;
        cur = prev;
    }
}

// ---------------------------------------------------------------------------
// fp32 -> bf16 conversion (n must be a multiple of 4)
// ---------------------------------------------------------------------------
__global__ void cvt_bf16_kernel(const float* __restrict__ in, __bf16* __restrict__ outp, int n)
{
    int i = (blockIdx.x * blockDim.x + threadIdx.x) * 4;
    if (i >= n) return;
    float4 f = *(const float4*)(in + i);
    outp[i + 0] = (__bf16)f.x;
    outp[i + 1] = (__bf16)f.y;
    outp[i + 2] = (__bf16)f.z;
    outp[i + 3] = (__bf16)f.w;
}

// ---------------------------------------------------------------------------
// WMMA GEMM:  D[M,Nout] = epilogue( A[M,K] @ W[Nout,K]^T + bias )
//   - A, W bf16 row-major; accumulate fp32 in v_wmma_f32_16x16x32_bf16
//   - act: 0 = identity, 1 = exact GELU
//   - resid: optional fp32 [M,Nout] added after activation
//   - C (fp32) and Cbf (bf16) outputs each optional (fused downstream feed)
// Each wave owns a 16x64 tile: 4 accumulators, K-loop of 16x16x32 WMMAs.
// Requires: M % 16 == 0, K % 32 == 0, Nout % 64 == 0.
// ---------------------------------------------------------------------------
__global__ __launch_bounds__(256)
void wmma_gemm_kernel(const __bf16* __restrict__ A,
                      const __bf16* __restrict__ W,
                      const float* __restrict__ bias,
                      const float* __restrict__ resid,
                      float* __restrict__ C,
                      __bf16* __restrict__ Cbf,
                      int M, int K, int Nout, int act)
{
    const int lane = threadIdx.x & 31;
    const int wave = threadIdx.x >> 5;
    const int mt   = blockIdx.x * 8 + wave;      // 16-row tile index (wave-uniform)
    const int m0   = mt * 16;
    if (m0 >= M) return;                          // uniform per wave: EXEC stays all-ones

    const int n0   = blockIdx.y * 64;
    const int half = lane >> 4;                   // 0 or 1
    const int l16  = lane & 15;
    const int mrow = m0 + l16;

    v8f acc0 = {}, acc1 = {}, acc2 = {}, acc3 = {};

    const __bf16* arow = A + (size_t)mrow * K;
    const __bf16* wrow = W + (size_t)(n0 + l16) * K;

    for (int kt = 0; kt < K; kt += 32) {
        // A fragment (16x32): elems 0..7 <- K=kt+half*8+(0..7); 8..15 <- +16
        const __bf16* ap = arow + kt + half * 8;
        __builtin_prefetch(ap + 32, 0, 0);        // next K tile -> global_prefetch
        V16 af;
        af.u[0] = *(const uint4*)(ap);
        af.u[1] = *(const uint4*)(ap + 16);

        // B fragments (32x16), B[k,n] = W[n,k]:
        // lane(n=l16, half): elems 0..7 <- K = kt + half*8 + (0..7); 8..15 <- +16
        #pragma unroll
        for (int nt = 0; nt < 4; ++nt) {
            const __bf16* wp = wrow + (size_t)nt * 16 * K + kt + half * 8;
            V16 bfrag;
            bfrag.u[0] = *(const uint4*)(wp);
            bfrag.u[1] = *(const uint4*)(wp + 16);

            v8f* accp = (nt == 0) ? &acc0 : (nt == 1) ? &acc1 : (nt == 2) ? &acc2 : &acc3;
            *accp = __builtin_amdgcn_wmma_f32_16x16x32_bf16(
                        false, af.v, false, bfrag.v, (short)0, *accp, false, false);
        }
    }

    // Epilogue. C/D layout: VGPR j, lanes 0..15: M=j, lanes 16..31: M=8+j.
    #pragma unroll
    for (int nt = 0; nt < 4; ++nt) {
        const v8f acc = (nt == 0) ? acc0 : (nt == 1) ? acc1 : (nt == 2) ? acc2 : acc3;
        const int col = n0 + nt * 16 + l16;
        const float bv = bias ? bias[col] : 0.0f;
        #pragma unroll
        for (int j = 0; j < 8; ++j) {
            const size_t row = (size_t)(m0 + half * 8 + j);
            float v = acc[j] + bv;
            if (act == 1) v = gelu_exact(v);
            if (resid) v += resid[row * Nout + col];
            if (C)   C[row * Nout + col] = v;
            if (Cbf) Cbf[row * Nout + col] = (__bf16)v;
        }
    }
}

// ---------------------------------------------------------------------------
// Attention helpers
// ---------------------------------------------------------------------------
__global__ void init_attn_kernel(float* amax, float* denom, float* agg, int N)
{
    int i = blockIdx.x * blockDim.x + threadIdx.x;
    if (i < N * NHEAD) { amax[i] = -1e30f; denom[i] = 0.0f; }
    if (i < N * HID)   { agg[i] = 0.0f; }
}

__global__ void edge_alpha_kernel(const float* __restrict__ q, const float* __restrict__ k,
                                  const int* __restrict__ src, const int* __restrict__ dst,
                                  float* __restrict__ alpha, float* __restrict__ amax, int E)
{
    int idx = blockIdx.x * blockDim.x + threadIdx.x;
    if (idx >= E * NHEAD) return;
    int e = idx >> 3, hh = idx & 7;
    int s = src[e], d = dst[e];
    const float* qp = q + (size_t)d * HID + hh * DHEAD;
    const float* kp = k + (size_t)s * HID + hh * DHEAD;
    float acc = 0.0f;
    #pragma unroll
    for (int i = 0; i < DHEAD; ++i) acc += qp[i] * kp[i];
    acc *= 0.25f;                                  // 1/sqrt(16)
    alpha[idx] = acc;
    atomicMaxF(&amax[(size_t)d * NHEAD + hh], acc);
}

__global__ void edge_exp_kernel(float* __restrict__ alpha, const float* __restrict__ amax,
                                float* __restrict__ denom, const int* __restrict__ dst, int E)
{
    int idx = blockIdx.x * blockDim.x + threadIdx.x;
    if (idx >= E * NHEAD) return;
    int e = idx >> 3, hh = idx & 7;
    int d = dst[e];
    float ex = expf(alpha[idx] - amax[(size_t)d * NHEAD + hh]);
    alpha[idx] = ex;
    atomicAdd(&denom[(size_t)d * NHEAD + hh], ex);
}

__global__ void edge_agg_kernel(const float* __restrict__ alpha, const float* __restrict__ denom,
                                const float* __restrict__ v, const int* __restrict__ src,
                                const int* __restrict__ dst, float* __restrict__ agg, int E)
{
    int idx = blockIdx.x * blockDim.x + threadIdx.x;
    if (idx >= E * NHEAD) return;
    int e = idx >> 3, hh = idx & 7;
    int s = src[e], d = dst[e];
    float a = alpha[idx] / (denom[(size_t)d * NHEAD + hh] + 1e-16f);
    const float* vp = v + (size_t)s * HID + hh * DHEAD;
    float* op = agg + (size_t)d * HID + hh * DHEAD;
    #pragma unroll
    for (int i = 0; i < DHEAD; ++i) atomicAdd(&op[i], a * vp[i]);
}

// ---------------------------------------------------------------------------
// Fused: beta-gate (sigmoid over [out, x_r, out-x_r] @ Wbeta^T), residual add,
// LayerNorm. One wave32 per node; 4 elements per lane; __shfl_xor reductions.
// Writes fp32 hout and bf16 houtbf (feeds the FFN's WMMA GEMM).
// ---------------------------------------------------------------------------
__global__ __launch_bounds__(256)
void beta_ln_kernel(const float* __restrict__ agg, const float* __restrict__ sb,
                    const float* __restrict__ hin, const float* __restrict__ Wbeta,
                    const float* __restrict__ g, const float* __restrict__ b,
                    float* __restrict__ hout, __bf16* __restrict__ houtbf, int N)
{
    int lane = threadIdx.x & 31;
    int node = (blockIdx.x << 3) + (threadIdx.x >> 5);
    if (node >= N) return;
    size_t base = (size_t)node * HID;

    float o[4], xr[4], hv[4];
    float dot = 0.0f;
    #pragma unroll
    for (int i = 0; i < 4; ++i) {
        int c = i * 32 + lane;
        o[i]  = agg[base + c];
        xr[i] = sb[base + c];
        hv[i] = hin[base + c];
        dot += o[i] * Wbeta[c] + xr[i] * Wbeta[HID + c] + (o[i] - xr[i]) * Wbeta[2 * HID + c];
    }
    #pragma unroll
    for (int m = 16; m >= 1; m >>= 1) dot += __shfl_xor(dot, m, 32);
    float beta = 1.0f / (1.0f + expf(-dot));

    float y[4];
    float s1 = 0.0f;
    #pragma unroll
    for (int i = 0; i < 4; ++i) {
        y[i] = beta * xr[i] + (1.0f - beta) * o[i] + hv[i];  // gate + residual
        s1 += y[i];
    }
    #pragma unroll
    for (int m = 16; m >= 1; m >>= 1) s1 += __shfl_xor(s1, m, 32);
    float mean = s1 * (1.0f / HID);

    float s2 = 0.0f;
    #pragma unroll
    for (int i = 0; i < 4; ++i) { float d = y[i] - mean; s2 += d * d; }
    #pragma unroll
    for (int m = 16; m >= 1; m >>= 1) s2 += __shfl_xor(s2, m, 32);
    float inv = rsqrtf(s2 * (1.0f / HID) + 1e-5f);

    #pragma unroll
    for (int i = 0; i < 4; ++i) {
        int c = i * 32 + lane;
        float r = (y[i] - mean) * inv * g[c] + b[c];
        hout[base + c] = r;
        houtbf[base + c] = (__bf16)r;
    }
}

// ---------------------------------------------------------------------------
// Host-side orchestration
// ---------------------------------------------------------------------------
static inline void launch_gemm(const __bf16* A, const __bf16* W, const float* bias,
                               const float* resid, float* C, __bf16* Cbf,
                               int M, int K, int Nout, int act, hipStream_t s)
{
    dim3 grid((M / 16 + 7) / 8, Nout / 64);
    wmma_gemm_kernel<<<grid, 256, 0, s>>>(A, W, bias, resid, C, Cbf, M, K, Nout, act);
}

static inline void launch_cvt(const float* in, __bf16* outp, long long n, hipStream_t s)
{
    int blocks = (int)((n / 4 + 255) / 256);
    cvt_bf16_kernel<<<blocks, 256, 0, s>>>(in, outp, (int)n);
}

extern "C" void kernel_launch(void* const* d_in, const int* in_sizes, int n_in,
                              void* d_out, int out_size, void* d_ws, size_t ws_size,
                              hipStream_t stream)
{
    const float* x     = (const float*)d_in[0];
    const int*   ei    = (const int*)  d_in[1];
    const float* Wi    = (const float*)d_in[2];
    const float* bi    = (const float*)d_in[3];
    const float* Wq    = (const float*)d_in[4];
    const float* bq    = (const float*)d_in[5];
    const float* Wk    = (const float*)d_in[6];
    const float* bk    = (const float*)d_in[7];
    const float* Wv    = (const float*)d_in[8];
    const float* bv    = (const float*)d_in[9];
    const float* Ws    = (const float*)d_in[10];
    const float* bs    = (const float*)d_in[11];
    const float* Wbeta = (const float*)d_in[12];
    const float* lng   = (const float*)d_in[13];
    const float* lnb   = (const float*)d_in[14];
    const float* W1    = (const float*)d_in[15];
    const float* b1    = (const float*)d_in[16];
    const float* W2    = (const float*)d_in[17];
    const float* b2    = (const float*)d_in[18];
    const float* Wo    = (const float*)d_in[19];
    const float* bo    = (const float*)d_in[20];

    const int N = in_sizes[0] / 64;      // 50000
    const int E = in_sizes[1] / 2;       // 600000
    const int L = 4;
    const int IN_C = 64, OUT_C = 64;
    const int* src = ei;
    const int* dst = ei + E;

    // ---- Workspace carve-up: fp32 region, then bf16 region ----
    float* ws = (float*)d_ws;
    size_t o = 0;
    float* hA    = ws + o; o += (size_t)N * HID;     // fp32 h (residual feed)
    float* hB    = ws + o; o += (size_t)N * HID;
    float* qb    = ws + o; o += (size_t)N * HID;
    float* kb    = ws + o; o += (size_t)N * HID;
    float* vb    = ws + o; o += (size_t)N * HID;
    float* sb    = ws + o; o += (size_t)N * HID;
    float* agg   = ws + o; o += (size_t)N * HID;
    float* alpha = ws + o; o += (size_t)E * NHEAD;
    float* amax  = ws + o; o += (size_t)N * NHEAD;
    float* denom = ws + o; o += (size_t)N * NHEAD;

    __bf16* bws = (__bf16*)(ws + o);
    size_t ob = 0;
    __bf16* xbf   = bws + ob; ob += (size_t)N * IN_C;
    __bf16* hAbf  = bws + ob; ob += (size_t)N * HID;
    __bf16* hBbf  = bws + ob; ob += (size_t)N * HID;
    __bf16* midbf = bws + ob; ob += (size_t)N * 4 * HID;
    __bf16* Wibf  = bws + ob; ob += (size_t)HID * IN_C;
    __bf16* Wqbf  = bws + ob; ob += (size_t)L * HID * HID;
    __bf16* Wkbf  = bws + ob; ob += (size_t)L * HID * HID;
    __bf16* Wvbf  = bws + ob; ob += (size_t)L * HID * HID;
    __bf16* Wsbf  = bws + ob; ob += (size_t)L * HID * HID;
    __bf16* W1bf  = bws + ob; ob += (size_t)L * 4 * HID * HID;
    __bf16* W2bf  = bws + ob; ob += (size_t)L * HID * 4 * HID;
    __bf16* Wobf  = bws + ob; ob += (size_t)OUT_C * HID;

    // ---- One-time conversions (weights + input x) ----
    launch_cvt(x,  xbf,  (long long)N * IN_C, stream);
    launch_cvt(Wi, Wibf, (long long)HID * IN_C, stream);
    launch_cvt(Wq, Wqbf, (long long)L * HID * HID, stream);
    launch_cvt(Wk, Wkbf, (long long)L * HID * HID, stream);
    launch_cvt(Wv, Wvbf, (long long)L * HID * HID, stream);
    launch_cvt(Ws, Wsbf, (long long)L * HID * HID, stream);
    launch_cvt(W1, W1bf, (long long)L * 4 * HID * HID, stream);
    launch_cvt(W2, W2bf, (long long)L * HID * 4 * HID, stream);
    launch_cvt(Wo, Wobf, (long long)OUT_C * HID, stream);

    const int eb = 256;
    const int egrid = (E * NHEAD + eb - 1) / eb;
    const int igrid = (N * HID + eb - 1) / eb;
    const int ngrid = (N + 7) / 8;

    // h = gelu(x @ Wi.T + bi)  -> hA (fp32, residual) + hAbf (GEMM feed)
    launch_gemm(xbf, Wibf, bi, nullptr, hA, hAbf, N, IN_C, HID, 1, stream);

    for (int l = 0; l < L; ++l) {
        const __bf16* Wql = Wqbf + (size_t)l * HID * HID;
        const __bf16* Wkl = Wkbf + (size_t)l * HID * HID;
        const __bf16* Wvl = Wvbf + (size_t)l * HID * HID;
        const __bf16* Wsl = Wsbf + (size_t)l * HID * HID;
        const __bf16* W1l = W1bf + (size_t)l * 4 * HID * HID;
        const __bf16* W2l = W2bf + (size_t)l * HID * 4 * HID;

        launch_gemm(hAbf, Wql, bq + l * HID, nullptr, qb, nullptr, N, HID, HID, 0, stream);
        launch_gemm(hAbf, Wkl, bk + l * HID, nullptr, kb, nullptr, N, HID, HID, 0, stream);
        launch_gemm(hAbf, Wvl, bv + l * HID, nullptr, vb, nullptr, N, HID, HID, 0, stream);
        launch_gemm(hAbf, Wsl, bs + l * HID, nullptr, sb, nullptr, N, HID, HID, 0, stream);

        init_attn_kernel<<<igrid, eb, 0, stream>>>(amax, denom, agg, N);
        edge_alpha_kernel<<<egrid, eb, 0, stream>>>(qb, kb, src, dst, alpha, amax, E);
        edge_exp_kernel<<<egrid, eb, 0, stream>>>(alpha, amax, denom, dst, E);
        edge_agg_kernel<<<egrid, eb, 0, stream>>>(alpha, denom, vb, src, dst, agg, E);

        // beta gate + residual + LayerNorm -> hB (fp32) + hBbf (GEMM feed)
        beta_ln_kernel<<<ngrid, 256, 0, stream>>>(agg, sb, hA, Wbeta + (size_t)l * 3 * HID,
                                                  lng + l * HID, lnb + l * HID, hB, hBbf, N);

        // FFN: mid = gelu(hB @ W1.T + b1) (bf16 only); hA = mid @ W2.T + b2 + hB
        launch_gemm(hBbf, W1l, b1 + l * 4 * HID, nullptr, nullptr, midbf, N, HID, 4 * HID, 1, stream);
        launch_gemm(midbf, W2l, b2 + l * HID, hB, hA, hAbf, N, 4 * HID, HID, 0, stream);
    }

    // out = h @ Wo.T + bo
    launch_gemm(hAbf, Wobf, bo, nullptr, (float*)d_out, nullptr, N, HID, OUT_C, 0, stream);
}
